// AntiAliasedUpsampleVocoderUpsampleBlock_45303315038320
// MI455X (gfx1250) — matrix operand order repaired and checked by reading, not compiled, not run on hardware
//
#include <hip/hip_runtime.h>
#include <hip/hip_bf16.h>

typedef __attribute__((ext_vector_type(16))) __bf16 v16bf;
typedef __attribute__((ext_vector_type(8)))  __bf16 v8bf;
typedef __attribute__((ext_vector_type(8)))  float  v8f;

#define C_IN   512
#define C_OUT  256
#define KS     7
#define T_IN   4096
#define UPF    4
#define T_OUT  (T_IN * UPF)      // 16384
#define KDIM   (C_IN * KS)       // 3584
#define NT     128               // time tile per block
#define NTHREADS 512             // 16 waves
#define TROWS  (NT + KS - 1)     // 134 rows of upsampled tile in LDS
#define LDS_W  (C_IN + 8)        // 520: pad rows (1040 B) to kill LDS bank conflicts

// ---------------------------------------------------------------------------
// Kernel 1: weight normalization  w[m][kk*512+ci] = bf16(g[m]*v[m][ci][kk]/||v[m]||)
// ---------------------------------------------------------------------------
__global__ __launch_bounds__(256) void prep_weights(
    const float* __restrict__ v, const float* __restrict__ g,
    __bf16* __restrict__ wout)
{
    const int m = blockIdx.x;
    const float* vm = v + m * KDIM;
    float ss = 0.f;
    for (int i = threadIdx.x; i < KDIM; i += 256) { float t = vm[i]; ss += t * t; }
    for (int off = 16; off > 0; off >>= 1) ss += __shfl_down(ss, off, 32);
    __shared__ float red[8];
    if ((threadIdx.x & 31) == 0) red[threadIdx.x >> 5] = ss;
    __syncthreads();
    if (threadIdx.x == 0) {
        float tot = 0.f;
        #pragma unroll
        for (int i = 0; i < 8; ++i) tot += red[i];
        red[0] = tot;
    }
    __syncthreads();
    const float scale = g[m] * rsqrtf(red[0]);
    for (int i = threadIdx.x; i < KDIM; i += 256) {
        int ci = i / KS, kk = i - ci * KS;          // conv_v layout [m][ci][kk]
        wout[m * KDIM + kk * C_IN + ci] = (__bf16)(vm[i] * scale);
    }
}

// ---------------------------------------------------------------------------
// Kernel 2: fused polyphase-upsample + bf16 WMMA GEMM conv + bias + snake
// block = (time tile of 128, batch); 512 threads = 16 waves
// wave w: M-group (w&7) -> m tiles {(w&7)*16, (w&7)*16+128}; N-group (w>>3) -> 64 cols
// ---------------------------------------------------------------------------
__global__ __launch_bounds__(NTHREADS) void upsample_gemm_snake(
    const float*  __restrict__ x,       // [16][512][4096]
    const float*  __restrict__ lowpass, // [13]
    const __bf16* __restrict__ W,       // [256][3584], k-reordered
    const float*  __restrict__ bias,    // [256]
    const float*  __restrict__ alpha,   // [256]
    float*        __restrict__ out)     // [16][256][16384]
{
    __shared__ __bf16 tile[TROWS * LDS_W];   // ~139 KB
    __shared__ float  slp[16];

    const int b   = blockIdx.y;
    const int t0  = blockIdx.x * NT;
    const int tid = threadIdx.x;

    if (tid < 13) slp[tid] = lowpass[tid];
    __syncthreads();

    // ---- prologue: polyphase (zero-stuff x4 + 13-tap lowpass) into LDS ----
    // xu[t] = 4 * sum_{j: (t-6+j)%4==0} k[j] * x[(t-6+j)/4]
    for (int i = tid; i < TROWS * C_IN; i += NTHREADS) {
        const int tloc = i >> 9;              // / 512
        const int c    = i & (C_IN - 1);
        const int t    = t0 - 3 + tloc;       // global upsampled time
        const int s    = t - 6;
        const int r    = (4 - ((s + 12) & 3)) & 3;   // first contributing tap
        const int tin  = (s + r) >> 2;               // input sample of first tap
        const float* xc = x + ((b << 9) + c) * T_IN;
        float acc = 0.f;
        int ti = tin;
        if ((unsigned)ti < (unsigned)T_IN) acc += slp[r]     * xc[ti];
        ti = tin + 1;
        if ((unsigned)ti < (unsigned)T_IN) acc += slp[r + 4] * xc[ti];
        ti = tin + 2;
        if ((unsigned)ti < (unsigned)T_IN) acc += slp[r + 8] * xc[ti];
        if (r == 0) {
            ti = tin + 3;
            if ((unsigned)ti < (unsigned)T_IN) acc += slp[12] * xc[ti];
        }
        tile[tloc * LDS_W + c] = (__bf16)(4.0f * acc);
    }
    __syncthreads();

    // ---- main loop: W(256x3584) @ Xcol(3584x128) via v_wmma_f32_16x16x32_bf16
    const int wave = tid >> 5;
    const int lane = tid & 31;
    const int lh   = lane & 15;   // m (A) / n (B,C) within a 16x16 tile
    const int hl   = lane >> 4;   // half-wave select
    const int mgrp = wave & 7;
    const int nbase = (wave >> 3) * 64;   // N-group column offset (0 or 64)

    v8f acc[2][4];
    #pragma unroll
    for (int mi = 0; mi < 2; ++mi)
        #pragma unroll
        for (int nt = 0; nt < 4; ++nt)
            acc[mi][nt] = (v8f){0.f,0.f,0.f,0.f,0.f,0.f,0.f,0.f};

    const int mbase0 = mgrp * 16;
    const int mbase1 = mgrp * 16 + 128;
    const __bf16* Wr0 = W + (mbase0 + lh) * KDIM;
    const __bf16* Wr1 = W + (mbase1 + lh) * KDIM;

    for (int kstep = 0; kstep < KDIM / 32; ++kstep) {   // 112 steps
        const int kk = kstep >> 4;          // filter tap (0..6)
        const int c0 = (kstep & 15) << 5;   // channel base (0..480)
        const int k0 = kstep << 5;          // flat kappa base

        // A tiles: lane half hl holds K = hl*8..+7 then 16+hl*8..+7
        v16bf a0, a1;
        *(v8bf*)&a0       = *(const v8bf*)(Wr0 + k0 + hl * 8);
        *((v8bf*)&a0 + 1) = *(const v8bf*)(Wr0 + k0 + 16 + hl * 8);
        *(v8bf*)&a1       = *(const v8bf*)(Wr1 + k0 + hl * 8);
        *((v8bf*)&a1 + 1) = *(const v8bf*)(Wr1 + k0 + 16 + hl * 8);

        #pragma unroll
        for (int nt = 0; nt < 4; ++nt) {
            // B tile: lane half hl holds channels hl*8..+7 then 16+hl*8..+7 at one time row
            const __bf16* bp =
                &tile[(nbase + nt * 16 + lh + kk) * LDS_W + c0 + hl * 8];
            v16bf bm;
            *(v8bf*)&bm       = *(const v8bf*)(bp);
            *((v8bf*)&bm + 1) = *(const v8bf*)(bp + 16);
            acc[0][nt] = __builtin_amdgcn_wmma_f32_16x16x32_bf16(
                false, a0, false, bm, (short)0, acc[0][nt], false, false);
            acc[1][nt] = __builtin_amdgcn_wmma_f32_16x16x32_bf16(
                false, a1, false, bm, (short)0, acc[1][nt], false, false);
        }
    }

    // ---- epilogue: bias + snake: y + sin(a*y)^2 / a ----
    #pragma unroll
    for (int mi = 0; mi < 2; ++mi) {
        const int mb = (mi == 0) ? mbase0 : mbase1;
        #pragma unroll
        for (int nt = 0; nt < 4; ++nt) {
            #pragma unroll
            for (int v = 0; v < 8; ++v) {
                const int m = mb + hl * 8 + v;
                const int t = t0 + nbase + nt * 16 + lh;
                float y = acc[mi][nt][v] + bias[m];
                float a = alpha[m];
                float sn = __sinf(a * y);
                out[((size_t)(b * C_OUT + m)) * T_OUT + t] = y + sn * sn / a;
            }
        }
    }
}

// ---------------------------------------------------------------------------
extern "C" void kernel_launch(void* const* d_in, const int* in_sizes, int n_in,
                              void* d_out, int out_size, void* d_ws, size_t ws_size,
                              hipStream_t stream) {
    (void)in_sizes; (void)n_in; (void)out_size; (void)ws_size;
    const float* x     = (const float*)d_in[0];
    const float* lp    = (const float*)d_in[1];
    const float* convv = (const float*)d_in[2];
    const float* convg = (const float*)d_in[3];
    const float* convb = (const float*)d_in[4];
    const float* alpha = (const float*)d_in[5];
    float* out = (float*)d_out;
    __bf16* W = (__bf16*)d_ws;   // 256*3584*2 = 1.75 MiB

    prep_weights<<<dim3(C_OUT), dim3(256), 0, stream>>>(convv, convg, W);
    upsample_gemm_snake<<<dim3(T_OUT / NT, 16), dim3(NTHREADS), 0, stream>>>(
        x, lp, W, convb, alpha, out);
}
